// SAFARILayer_5179730559542
// MI455X (gfx1250) — matrix operand order repaired and checked by reading, not compiled, not run on hardware
//
#include <hip/hip_runtime.h>
#include <hip/hip_bf16.h>

typedef __attribute__((ext_vector_type(16))) __bf16 v16bf;
typedef __attribute__((ext_vector_type(8)))  float  v8f;

#define B_TOT 512
#define T_LEN 64
#define I_FEAT 64
#define H_DIM 128
#define BT 64

static __device__ __forceinline__ v8f wmma_bf16(v16bf a, v16bf b, v8f c) {
  // emits v_wmma_f32_16x16x32_bf16
  return __builtin_amdgcn_wmma_f32_16x16x32_bf16(false, a, false, b, (short)0, c, false, false);
}

// A fragment (16x32 bf16, MxK) from LDS. Layout: lanes0-15: M=lane, halfs = K0..15;
// lanes16-31: halfs = K16..31. Rows padded to stride*2 bytes, 16B aligned.
static __device__ __forceinline__ v16bf load_a_lds(const __bf16* base, int stride,
                                                   int mbase, int kbase, int lane) {
  int row = mbase + (lane & 15);
  int k0  = kbase + ((lane >> 4) << 4);
  const __bf16* p = base + row * stride + k0;
  union { v16bf v; uint4 u[2]; } r;
  r.u[0] = *(const uint4*)(p);
  r.u[1] = *(const uint4*)(p + 8);
  return r.v;
}

// B fragment (32x16 bf16, KxN) gathered from a row-major K x N LDS buffer.
// Contiguous k order, same as load_a_lds: lanes0-15: N=lane, half j = K(kbase+j); lanes16-31: N=lane-16, half j = K(kbase+16+j).
static __device__ __forceinline__ v16bf load_b_lds(const __bf16* base, int stride,
                                                   int kbase, int nbase, int lane) {
  int col = nbase + (lane & 15);
  int k0  = kbase + ((lane >> 4) << 4);
  v16bf b;
#pragma unroll
  for (int j = 0; j < 16; ++j) b[j] = base[(k0 + j) * stride + col];
  return b;
}

// B fragment from a global row-major [N][K] f32 weight (torch Linear convention),
// i.e. B[k][n] = W[n][k]. Each lane reads 16 consecutive floats (4x global_load_b128).
static __device__ __forceinline__ v16bf load_b_w(const float* __restrict__ W, int ldk,
                                                 int nbase, int kbase, int lane) {
  int n  = nbase + (lane & 15);
  int k0 = kbase + ((lane >> 4) << 4);
  const float* p = W + (size_t)n * ldk + k0;
  v16bf b;
#pragma unroll
  for (int j = 0; j < 16; ++j) b[j] = (__bf16)p[j];
  return b;
}

static __device__ __forceinline__ float sigf(float x) { return 1.f / (1.f + __expf(-x)); }
static __device__ __forceinline__ float tanh_fast(float x) {
  float e = __expf(-2.f * x);
  return (1.f - e) / (1.f + e);
}

// ---------------------------------------------------------------------------
// Kernel 1: per-feature GRU over T=64 steps, fused with fp projection.
// grid = (I=64, B/64=8), block = 256 (8 waves). Wave w owns h-columns [16w,16w+16).
// ---------------------------------------------------------------------------
__global__ __launch_bounds__(256)
void gru_fp_kernel(const float* __restrict__ x, const float* __restrict__ W_ih,
                   const float* __restrict__ W_hh, const float* __restrict__ b_ih,
                   const float* __restrict__ b_hh, const float* __restrict__ fp_w,
                   const float* __restrict__ fp_b, float* __restrict__ ht_f32,
                   unsigned short* __restrict__ ht_b16) {
  __shared__ alignas(16) __bf16 sH[2][BT][136];   // ping-pong bf16 hidden state
  __shared__ alignas(16) float  sX[BT][68];       // x[btile][t] slice for feature i

  const int i    = blockIdx.x;
  const int b0   = blockIdx.y * BT;
  const int tid  = threadIdx.x;
  const int w    = tid >> 5;
  const int lane = tid & 31;
  const int col  = (w << 4) + (lane & 15);   // owned h-column (same for both half-waves)
  const int rsel = (lane >> 4) << 3;         // C-fragment row offset: 0 or 8

  __builtin_prefetch(W_hh + (size_t)i * 384 * H_DIM, 0, 0);

  // stage x[b0:b0+64, :, i] into LDS
  for (int e = tid; e < BT * T_LEN; e += 256) {
    int row = e >> 6, t = e & 63;
    sX[row][t] = x[((size_t)(b0 + row) * T_LEN + t) * I_FEAT + i];
  }
  // h0 = 0
  for (int e = tid; e < BT * 136; e += 256) (&sH[0][0][0])[e] = (__bf16)0.f;

  // per-lane gate scalars for owned column (gate order r,z,n at offsets 0,128,256)
  const size_t gi = (size_t)i * 384;
  float wih0 = W_ih[gi + col],       wih1 = W_ih[gi + 128 + col],  wih2 = W_ih[gi + 256 + col];
  float bih0 = b_ih[gi + col],       bih1 = b_ih[gi + 128 + col],  bih2 = b_ih[gi + 256 + col];
  float bhh0 = b_hh[gi + col],       bhh1 = b_hh[gi + 128 + col],  bhh2 = b_hh[gi + 256 + col];

  // W_hh^T B-fragments, resident in registers for the whole recurrence
  v16bf Bw[3][4];
#pragma unroll
  for (int g = 0; g < 3; ++g)
#pragma unroll
    for (int kt = 0; kt < 4; ++kt)
      Bw[g][kt] = load_b_w(W_hh + ((size_t)i * 384 + g * 128) * H_DIM, H_DIM,
                           w << 4, kt << 5, lane);

  float hreg[4][8];   // f32 master copy of owned h elements
#pragma unroll
  for (int m = 0; m < 4; ++m)
#pragma unroll
    for (int e = 0; e < 8; ++e) hreg[m][e] = 0.f;

  __syncthreads();

  for (int t = 0; t < T_LEN; ++t) {
    const int rb = t & 1, wb = rb ^ 1;
#pragma unroll
    for (int m = 0; m < 4; ++m) {
      v8f ar = {}, az = {}, an = {};
#pragma unroll
      for (int kt = 0; kt < 4; ++kt) {
        v16bf a = load_a_lds(&sH[rb][0][0], 136, m << 4, kt << 5, lane);
        ar = wmma_bf16(a, Bw[0][kt], ar);
        az = wmma_bf16(a, Bw[1][kt], az);
        an = wmma_bf16(a, Bw[2][kt], an);
      }
#pragma unroll
      for (int e = 0; e < 8; ++e) {
        int row   = (m << 4) + rsel + e;
        float xv  = sX[row][t];
        float gr  = ar[e] + bhh0, gz = az[e] + bhh1, gn = an[e] + bhh2;
        float r   = sigf(fmaf(xv, wih0, bih0) + gr);
        float z   = sigf(fmaf(xv, wih1, bih1) + gz);
        float n   = tanh_fast(fmaf(xv, wih2, bih2) + r * gn);
        float hn  = fmaf(z, hreg[m][e] - n, n);   // (1-z)*n + z*h
        hreg[m][e] = hn;
        sH[wb][row][col] = (__bf16)hn;
      }
    }
    __syncthreads();
  }

  // fused fp projection: ht = hT @ fp_w^T + fp_b  (final state is in sH[0])
  v16bf Bfp[4];
#pragma unroll
  for (int kt = 0; kt < 4; ++kt) Bfp[kt] = load_b_w(fp_w, H_DIM, w << 4, kt << 5, lane);
  float fb = fp_b[col];
#pragma unroll
  for (int m = 0; m < 4; ++m) {
    v8f acc = {};
#pragma unroll
    for (int kt = 0; kt < 4; ++kt) {
      v16bf a = load_a_lds(&sH[0][0][0], 136, m << 4, kt << 5, lane);
      acc = wmma_bf16(a, Bfp[kt], acc);
    }
#pragma unroll
    for (int e = 0; e < 8; ++e) {
      int row    = (m << 4) + rsel + e;
      float v    = acc[e] + fb;
      size_t idx = ((size_t)(b0 + row) * I_FEAT + i) * H_DIM + col;
      ht_f32[idx] = v;
      ht_b16[idx] = __builtin_bit_cast(unsigned short, (__bf16)v);
    }
  }
}

// ---------------------------------------------------------------------------
// Kernel 2: per-batch graph conv (adj@ht -> g1 -> adj@g -> g2) + QKV attention.
// grid = B, block = 256 (8 waves).
// ---------------------------------------------------------------------------
__global__ __launch_bounds__(256)
void post_kernel(const unsigned short* __restrict__ ht_b16, const float* __restrict__ adj,
                 const float* __restrict__ g1_w, const float* __restrict__ g1_b,
                 const float* __restrict__ g2_w, const float* __restrict__ g2_b,
                 const float* __restrict__ wq_w, const float* __restrict__ wq_b,
                 const float* __restrict__ wk_w, const float* __restrict__ wk_b,
                 const float* __restrict__ wv_w, const float* __restrict__ wv_b,
                 const float* __restrict__ out0_w, const float* __restrict__ out0_b,
                 float* __restrict__ out) {
  __shared__ alignas(16) __bf16 sA[I_FEAT][136];
  __shared__ alignas(16) __bf16 sB[I_FEAT][136];
  __shared__ alignas(16) __bf16 sAdj[I_FEAT][72];
  __shared__ float sQ[128], sP[128], sS[128], sW[128], sE[64], sRed[2];

  const int b    = blockIdx.x;
  const int tid  = threadIdx.x;
  const int w    = tid >> 5;
  const int lane = tid & 31;
  const int col  = (w << 4) + (lane & 15);
  const int rsel = (lane >> 4) << 3;

  for (int e = tid; e < I_FEAT * H_DIM; e += 256) {
    int row = e >> 7, c = e & 127;
    sA[row][c] = __builtin_bit_cast(__bf16, ht_b16[((size_t)b * I_FEAT + row) * H_DIM + c]);
  }
  for (int e = tid; e < I_FEAT * I_FEAT; e += 256) {
    int row = e >> 6, c = e & 63;
    sAdj[row][c] = (__bf16)adj[row * I_FEAT + c];   // 0/1, exact in bf16
  }
  __syncthreads();

  // t1 = adj @ ht  -> sB
#pragma unroll
  for (int m = 0; m < 4; ++m) {
    v8f acc = {};
#pragma unroll
    for (int kt = 0; kt < 2; ++kt) {
      v16bf a  = load_a_lds(&sAdj[0][0], 72, m << 4, kt << 5, lane);
      v16bf bf = load_b_lds(&sA[0][0], 136, kt << 5, w << 4, lane);
      acc = wmma_bf16(a, bf, acc);
    }
#pragma unroll
    for (int e = 0; e < 8; ++e) sB[(m << 4) + rsel + e][col] = (__bf16)acc[e];
  }
  __syncthreads();

  // g = relu(t1 @ g1^T + b1) -> sA
  {
    v16bf Bg[4];
#pragma unroll
    for (int kt = 0; kt < 4; ++kt) Bg[kt] = load_b_w(g1_w, H_DIM, w << 4, kt << 5, lane);
    float bias = g1_b[col];
#pragma unroll
    for (int m = 0; m < 4; ++m) {
      v8f acc = {};
#pragma unroll
      for (int kt = 0; kt < 4; ++kt)
        acc = wmma_bf16(load_a_lds(&sB[0][0], 136, m << 4, kt << 5, lane), Bg[kt], acc);
#pragma unroll
      for (int e = 0; e < 8; ++e)
        sA[(m << 4) + rsel + e][col] = (__bf16)fmaxf(acc[e] + bias, 0.f);
    }
  }
  __syncthreads();

  // t2 = adj @ g -> sB
#pragma unroll
  for (int m = 0; m < 4; ++m) {
    v8f acc = {};
#pragma unroll
    for (int kt = 0; kt < 2; ++kt) {
      v16bf a  = load_a_lds(&sAdj[0][0], 72, m << 4, kt << 5, lane);
      v16bf bf = load_b_lds(&sA[0][0], 136, kt << 5, w << 4, lane);
      acc = wmma_bf16(a, bf, acc);
    }
#pragma unroll
    for (int e = 0; e < 8; ++e) sB[(m << 4) + rsel + e][col] = (__bf16)acc[e];
  }
  __syncthreads();

  // ctx = relu(t2 @ g2^T + b2) -> sA
  {
    v16bf Bg[4];
#pragma unroll
    for (int kt = 0; kt < 4; ++kt) Bg[kt] = load_b_w(g2_w, H_DIM, w << 4, kt << 5, lane);
    float bias = g2_b[col];
#pragma unroll
    for (int m = 0; m < 4; ++m) {
      v8f acc = {};
#pragma unroll
      for (int kt = 0; kt < 4; ++kt)
        acc = wmma_bf16(load_a_lds(&sB[0][0], 136, m << 4, kt << 5, lane), Bg[kt], acc);
#pragma unroll
      for (int e = 0; e < 8; ++e)
        sA[(m << 4) + rsel + e][col] = (__bf16)fmaxf(acc[e] + bias, 0.f);
    }
  }
  __syncthreads();

  // ---- attention (algebraically collapsed; VALU) ----
  if (tid < 128) {               // q = ctx[last] @ wq^T + bq
    float acc = wq_b[tid];
    for (int k = 0; k < 128; ++k) acc = fmaf((float)sA[63][k], wq_w[tid * 128 + k], acc);
    sQ[tid] = acc;
  }
  __syncthreads();
  if (tid < 128) {               // p_k = sum_h q_h * wk[h][k]
    float acc = 0.f;
    for (int h = 0; h < 128; ++h) acc = fmaf(sQ[h], wk_w[h * 128 + tid], acc);
    sP[tid] = acc;
  }
  if (tid == 255) {              // e-bias = sum_h q_h * bk_h
    float acc = 0.f;
    for (int h = 0; h < 128; ++h) acc = fmaf(sQ[h], wk_b[h], acc);
    sRed[0] = acc;
  }
  __syncthreads();
  if (tid < 64) {                // e_i = ctx_i . p + e-bias
    float acc = sRed[0];
    for (int k = 0; k < 128; ++k) acc = fmaf((float)sA[tid][k], sP[k], acc);
    sE[tid] = acc;
  }
  __syncthreads();
  if (tid == 0) {                // softmax max
    float mx = sE[0];
    for (int j = 1; j < 64; ++j) mx = fmaxf(mx, sE[j]);
    sRed[0] = mx;
  }
  __syncthreads();
  if (tid < 64) sE[tid] = __expf(sE[tid] - sRed[0]);
  __syncthreads();
  if (tid == 0) {
    float s = 0.f;
    for (int j = 0; j < 64; ++j) s += sE[j];
    sRed[1] = 1.f / s;
  }
  __syncthreads();
  if (tid < 64) sE[tid] *= sRed[1];
  __syncthreads();
  if (tid < 128) {               // s_k = sum_i a_i * ctx[i][k]
    float acc = 0.f;
    for (int j = 0; j < 64; ++j) acc = fmaf(sE[j], (float)sA[j][tid], acc);
    sS[tid] = acc;
  }
  __syncthreads();
  if (tid < 128) {               // wctx_h = s . wv[h] + bv_h   (sum a = 1)
    float acc = wv_b[tid];
    for (int k = 0; k < 128; ++k) acc = fmaf(sS[k], wv_w[tid * 128 + k], acc);
    sW[tid] = acc;
  }
  __syncthreads();
  if (tid < 128) {               // out = relu(wctx @ out0^T + b)
    float acc = out0_b[tid];
    for (int h = 0; h < 128; ++h) acc = fmaf(sW[h], out0_w[tid * 128 + h], acc);
    out[(size_t)b * 128 + tid] = fmaxf(acc, 0.f);
  }
}

extern "C" void kernel_launch(void* const* d_in, const int* in_sizes, int n_in,
                              void* d_out, int out_size, void* d_ws, size_t ws_size,
                              hipStream_t stream) {
  (void)in_sizes; (void)n_in; (void)out_size; (void)ws_size;
  const float* x      = (const float*)d_in[0];
  const float* W_ih   = (const float*)d_in[1];
  const float* W_hh   = (const float*)d_in[2];
  const float* b_ih   = (const float*)d_in[3];
  const float* b_hh   = (const float*)d_in[4];
  const float* fp_w   = (const float*)d_in[5];
  const float* fp_b   = (const float*)d_in[6];
  const float* g1_w   = (const float*)d_in[7];
  const float* g1_b   = (const float*)d_in[8];
  const float* g2_w   = (const float*)d_in[9];
  const float* g2_b   = (const float*)d_in[10];
  const float* wq_w   = (const float*)d_in[11];
  const float* wq_b   = (const float*)d_in[12];
  const float* wk_w   = (const float*)d_in[13];
  const float* wk_b   = (const float*)d_in[14];
  const float* wv_w   = (const float*)d_in[15];
  const float* wv_b   = (const float*)d_in[16];
  const float* out0_w = (const float*)d_in[17];
  const float* out0_b = (const float*)d_in[18];
  const float* adj    = (const float*)d_in[19];

  float* out    = (float*)d_out;                       // [512,128]
  float* ht_f32 = out + (size_t)B_TOT * H_DIM;         // [512,64,128]
  unsigned short* ht_b16 = (unsigned short*)d_ws;      // 8 MB bf16 staging

  gru_fp_kernel<<<dim3(I_FEAT, B_TOT / BT), 256, 0, stream>>>(
      x, W_ih, W_hh, b_ih, b_hh, fp_w, fp_b, ht_f32, ht_b16);
  post_kernel<<<dim3(B_TOT), 256, 0, stream>>>(
      ht_b16, adj, g1_w, g1_b, g2_w, g2_b, wq_w, wq_b, wk_w, wk_b,
      wv_w, wv_b, out0_w, out0_b, out);
}
